// SelfAttention_40458591928776
// MI455X (gfx1250) — matrix operand order repaired and checked by reading, not compiled, not run on hardware
//
#include <hip/hip_runtime.h>
#include <hip/hip_bf16.h>

// ---------------------------------------------------------------------------
// Self-attention pipeline for MI455X (gfx1250, wave32, WMMA).
//   B=4, S=1024, D=2048, H=16, DH=128
// All matmuls in bf16 via v_wmma_f32_16x16x32_bf16 (f32 accumulate).
// Tile staging: double-buffered GLOBAL_LOAD_ASYNC_TO_LDS_B128 (ASYNCcnt) for
// non-transposed tiles; register transpose for the others. One barrier per
// k-tile; async stage of tile t+1 overlaps WMMA compute of tile t.
// ---------------------------------------------------------------------------

typedef __bf16 bf16;
typedef __attribute__((ext_vector_type(16))) __bf16 v16bf;
typedef __attribute__((ext_vector_type(8)))  float  v8f;

#define WMMA_BF16(a, b, c)                                                   \
  __builtin_amdgcn_wmma_f32_16x16x32_bf16(false, (a), false, (b), (short)0,  \
                                          (c), false, false)

// Async global->LDS copy of 16 bytes (GVS mode: SGPR base + 32-bit VGPR
// byte offset). ldsaddr is the LDS byte address (low 32 bits of the flat
// address of a __shared__ object, per the aperture mapping).
static __device__ __forceinline__ void async_copy_b128(uint32_t ldsaddr,
                                                       uint32_t gbyteoff,
                                                       const void* base) {
  asm volatile("global_load_async_to_lds_b128 %0, %1, %2"
               :
               : "v"(ldsaddr), "v"(gbyteoff), "s"(base)
               : "memory");
}
static __device__ __forceinline__ void async_wait0() {
  asm volatile("s_wait_asynccnt 0x0" ::: "memory");
}
static __device__ __forceinline__ uint32_t lds_addr_of(const void* p) {
  return (uint32_t)(uintptr_t)p;  // LDS aperture: addr[31:0] = LDS offset
}

static __device__ __forceinline__ v16bf make_frag(uint4 lo, uint4 hi) {
  v16bf f;
  __builtin_memcpy(&f, &lo, 16);
  __builtin_memcpy(((char*)&f) + 16, &hi, 16);
  return f;
}

static __device__ __forceinline__ float half_max16(float v) {
  // reduce across the 16-lane half (xor of lane bits 0..3 stays in the half)
  v = fmaxf(v, __shfl_xor(v, 1, 32));
  v = fmaxf(v, __shfl_xor(v, 2, 32));
  v = fmaxf(v, __shfl_xor(v, 4, 32));
  v = fmaxf(v, __shfl_xor(v, 8, 32));
  return v;
}
static __device__ __forceinline__ float half_sum16(float v) {
  v += __shfl_xor(v, 1, 32);
  v += __shfl_xor(v, 2, 32);
  v += __shfl_xor(v, 4, 32);
  v += __shfl_xor(v, 8, 32);
  return v;
}

// ---------------------------------------------------------------------------
// 1) fp32 -> bf16 conversion (grid-stride)
// ---------------------------------------------------------------------------
__global__ __launch_bounds__(256) void f32_to_bf16_kernel(
    const float* __restrict__ in, bf16* __restrict__ out, size_t n) {
  size_t i = (size_t)blockIdx.x * blockDim.x + threadIdx.x;
  size_t stride = (size_t)gridDim.x * blockDim.x;
  for (; i < n; i += stride) out[i] = (bf16)in[i];
}

// ---------------------------------------------------------------------------
// 2) Tiled bf16 WMMA GEMM: C[M,N] = A[M,K] * B[K,N], A/B row-major bf16.
//    WG tile 128x256, 8 waves (2x4), wave tile 64x64 (4x4 WMMA 16x16 tiles).
//    Double-buffered: async A stage + register-transposed B stage of tile
//    t+1 overlap WMMA compute of tile t.
// ---------------------------------------------------------------------------
template <bool OUT_F32>
__global__ __launch_bounds__(256) void gemm_wmma_bf16(
    const bf16* __restrict__ A, const bf16* __restrict__ B,
    void* __restrict__ C, int M, int N, int K) {
  constexpr int BM = 128, BN = 256, BK = 32, PAD = 8;
  __shared__ __align__(16) bf16 As[2][BM][BK + PAD];   // (m, k)
  __shared__ __align__(16) bf16 BsT[2][BN][BK + PAD];  // (n, k) transposed

  const int tid  = threadIdx.x;
  const int lane = tid & 31;
  const int wv   = tid >> 5;
  const int hi   = lane >> 4;
  const int l16  = lane & 15;
  const int waveM = wv >> 2;  // 0..1
  const int waveN = wv & 3;   // 0..3
  const int m0 = blockIdx.y * BM;
  const int n0 = blockIdx.x * BN;

  v8f acc[4][4] = {};

  auto stage = [&](int k0, int buf) {
    // ---- A tile 128x32 via async copies (2 x b128 per thread) ----
    {
      int row = tid >> 1;
      int kc  = (tid & 1) * 16;
      uint32_t goff = (uint32_t)(((size_t)(m0 + row) * K + k0 + kc) * 2);
      async_copy_b128(lds_addr_of(&As[buf][row][kc]), goff, A);
      async_copy_b128(lds_addr_of(&As[buf][row][kc + 8]), goff + 16, A);
    }
    // ---- B tile transposed: 32x256 -> BsT[n][k] ----
    {
      int krow = tid >> 3;
      int nc   = (tid & 7) * 32;
      const bf16* src = B + (size_t)(k0 + krow) * N + n0 + nc;
#pragma unroll
      for (int c = 0; c < 4; ++c) {
        uint4 raw = *(const uint4*)(src + c * 8);
        bf16 tmp[8];
        __builtin_memcpy(tmp, &raw, 16);
#pragma unroll
        for (int j = 0; j < 8; ++j) BsT[buf][nc + c * 8 + j][krow] = tmp[j];
      }
    }
  };

  stage(0, 0);
  async_wait0();
  __syncthreads();

  const int nT = K / BK;
  for (int t = 0; t < nT; ++t) {
    const int cur = t & 1;
    if (t + 1 < nT) stage((t + 1) * BK, cur ^ 1);
    if (t + 2 < nT)
      __builtin_prefetch(
          B + (size_t)((t + 2) * BK + (tid >> 3)) * N + n0 + (tid & 7) * 32, 0, 0);

    // ---- fragments (ISA 7.12.2 layouts) ----
    v16bf afrag[4], bfrag[4];
#pragma unroll
    for (int i = 0; i < 4; ++i) {
      const bf16* ap = &As[cur][waveM * 64 + i * 16 + l16][0];
      // A: e=0..7 -> k = e + 8*hi ; e=8..15 -> k = 16 + (e-8) + 8*hi
      afrag[i] = make_frag(*(const uint4*)(ap + 8 * hi),
                           *(const uint4*)(ap + 16 + 8 * hi));
    }
#pragma unroll
    for (int j = 0; j < 4; ++j) {
      const bf16* bp = &BsT[cur][waveN * 64 + j * 16 + l16][0];
      // B: e=0..7 -> k = e + 8*hi ; e=8..15 -> k = 16 + (e-8) + 8*hi
      bfrag[j] = make_frag(*(const uint4*)(bp + 8 * hi),
                           *(const uint4*)(bp + 16 + 8 * hi));
    }
#pragma unroll
    for (int i = 0; i < 4; ++i)
#pragma unroll
      for (int j = 0; j < 4; ++j)
        acc[i][j] = WMMA_BF16(afrag[i], bfrag[j], acc[i][j]);

    async_wait0();
    __syncthreads();
  }

  // ---- epilogue: C layout row = r + 8*hi, col = l16 ----
#pragma unroll
  for (int i = 0; i < 4; ++i) {
#pragma unroll
    for (int j = 0; j < 4; ++j) {
      int col = n0 + waveN * 64 + j * 16 + l16;
#pragma unroll
      for (int r = 0; r < 8; ++r) {
        int row = m0 + waveM * 64 + i * 16 + r + 8 * hi;
        if (OUT_F32)
          ((float*)C)[(size_t)row * N + col] = acc[i][j][r];
        else
          ((bf16*)C)[(size_t)row * N + col] = (bf16)acc[i][j][r];
      }
    }
  }
}

// ---------------------------------------------------------------------------
// 3) Fused RMSNorm(q,k) + RoPE + repack to (B*H, S, DH) bf16.
//    One block per (b,s) row; 8 waves; wave w handles heads w and w+8.
//    Attention scale 1/sqrt(DH) folded into Q.
// ---------------------------------------------------------------------------
__global__ __launch_bounds__(256) void rmsnorm_rope_pack(
    const bf16* __restrict__ qkv, const float* __restrict__ pe,
    const float* __restrict__ qscale, const float* __restrict__ qbias,
    const float* __restrict__ kscale, const float* __restrict__ kbias,
    bf16* __restrict__ Qp, bf16* __restrict__ Kp, bf16* __restrict__ Vp) {
  const int row  = blockIdx.x;  // b*1024 + s
  const int wv   = threadIdx.x >> 5;
  const int lane = threadIdx.x & 31;
  const int d0   = lane * 4;
  const int b = row >> 10, s = row & 1023;
  const float ATT_SCALE = 0.08838834764831845f;  // 1/sqrt(128)
  const float* perow = pe + (size_t)row * 256;   // (64 pairs) x (2x2)

  for (int h = wv; h < 16; h += 8) {
    const size_t base  = (size_t)row * 6144 + (size_t)h * 128 + d0;
    const size_t pbase = ((size_t)(b * 16 + h) * 1024 + s) * 128 + d0;

#pragma unroll
    for (int which = 0; which < 2; ++which) {  // 0 = q, 1 = k
      const size_t src = base + (size_t)which * 2048;
      const float* sc = which ? kscale : qscale;
      const float* bi = which ? kbias  : qbias;
      float u[4];
#pragma unroll
      for (int i = 0; i < 4; ++i) u[i] = (float)qkv[src + i];
      float ss = u[0] * u[0] + u[1] * u[1] + u[2] * u[2] + u[3] * u[3];
#pragma unroll
      for (int off = 16; off > 0; off >>= 1) ss += __shfl_xor(ss, off, 32);
      float inv = rsqrtf(ss * (1.0f / 128.0f) + 1e-6f);
      float un[4];
#pragma unroll
      for (int i = 0; i < 4; ++i) un[i] = u[i] * inv * sc[d0 + i] + bi[d0 + i];

      bf16* dst = which ? Kp : Qp;
      float oscale = which ? 1.0f : ATT_SCALE;
#pragma unroll
      for (int pr = 0; pr < 2; ++pr) {
        int j = (d0 >> 1) + pr;
        float e0 = un[pr * 2], e1 = un[pr * 2 + 1];
        float r0 = perow[j * 4 + 0] * e0 + perow[j * 4 + 1] * e1;
        float r1 = perow[j * 4 + 2] * e0 + perow[j * 4 + 3] * e1;
        dst[pbase + pr * 2]     = (bf16)(r0 * oscale);
        dst[pbase + pr * 2 + 1] = (bf16)(r1 * oscale);
      }
    }
    // v: straight repack
#pragma unroll
    for (int i = 0; i < 4; ++i) Vp[pbase + i] = qkv[base + 4096 + i];
  }
}

// ---------------------------------------------------------------------------
// 4) Flash attention, WMMA bf16. Grid (16 query tiles, 64 bh). 4 waves/WG,
//    wave = 16 query rows. 32-key tiles; online softmax; P via LDS relayout.
//    K tile async-staged, V tile register-transposed; both double-buffered
//    so staging of tile t+1 overlaps score/softmax/PV compute of tile t.
// ---------------------------------------------------------------------------
__global__ __launch_bounds__(128) void flash_attn_wmma(
    const bf16* __restrict__ Qp, const bf16* __restrict__ Kp,
    const bf16* __restrict__ Vp, bf16* __restrict__ attn) {
  constexpr int SQ = 1024, DH = 128;
  __shared__ __align__(16) bf16 Ks[2][32][DH + 8];  // (key, dh) row-major
  __shared__ __align__(16) bf16 VsT[2][128][40];    // (dh, key) transposed
  __shared__ __align__(16) bf16 Ps[4][16][40];      // per-wave P (row, key)

  const int tid  = threadIdx.x;
  const int lane = tid & 31;
  const int wv   = tid >> 5;
  const int hi   = lane >> 4;
  const int l16  = lane & 15;
  const int qt = blockIdx.x;  // 0..15
  const int bh = blockIdx.y;  // 0..63
  const int b = bh >> 4, h = bh & 15;
  const int m0 = qt * 64 + wv * 16;

  // Q fragments for our 16 rows (held in VGPRs across the whole loop)
  v16bf qfrag[4];
  {
    const bf16* qrow = Qp + ((size_t)bh * SQ + m0 + l16) * DH;
#pragma unroll
    for (int kc = 0; kc < 4; ++kc)
      qfrag[kc] = make_frag(*(const uint4*)(qrow + kc * 32 + 8 * hi),
                            *(const uint4*)(qrow + kc * 32 + 16 + 8 * hi));
  }

  v8f acc[8] = {};
  float mstate[8], lstate[8];
#pragma unroll
  for (int r = 0; r < 8; ++r) { mstate[r] = -1e30f; lstate[r] = 0.f; }

  auto stage = [&](int kt, int buf) {
    // ---- async K tile (32 keys x 128 dh, row-major, padded rows) ----
    {
      int key = tid >> 2;        // 0..31
      int c0  = (tid & 3) * 32;  // element offset within the row
      uint32_t goff =
          (uint32_t)((((size_t)bh * SQ + kt * 32 + key) * DH + c0) * 2);
#pragma unroll
      for (int c = 0; c < 4; ++c)
        async_copy_b128(lds_addr_of(&Ks[buf][key][c0 + c * 8]), goff + c * 16,
                        Kp);
    }
    // ---- V tile transposed into LDS (32 keys x 128 dh) ----
    {
      int key = tid >> 2;
      int dh0 = (tid & 3) * 32;
      const bf16* vrow = Vp + ((size_t)bh * SQ + kt * 32 + key) * DH + dh0;
#pragma unroll
      for (int c = 0; c < 4; ++c) {
        uint4 raw = *(const uint4*)(vrow + c * 8);
        bf16 tmp[8];
        __builtin_memcpy(tmp, &raw, 16);
#pragma unroll
        for (int j = 0; j < 8; ++j) VsT[buf][dh0 + c * 8 + j][key] = tmp[j];
      }
    }
  };

  stage(0, 0);
  async_wait0();
  __syncthreads();

  constexpr int nKT = SQ / 32;
  for (int kt = 0; kt < nKT; ++kt) {
    const int cur = kt & 1;
    if (kt + 1 < nKT) stage(kt + 1, cur ^ 1);

    // ---- scores S = Q K^T (16 x 32), K fragments from LDS ----
    v8f sc[2] = {};
#pragma unroll
    for (int nt = 0; nt < 2; ++nt) {
      const bf16* krow = &Ks[cur][nt * 16 + l16][0];
#pragma unroll
      for (int kc = 0; kc < 4; ++kc) {
        v16bf kfrag = make_frag(*(const uint4*)(krow + kc * 32 + 8 * hi),
                                *(const uint4*)(krow + kc * 32 + 16 + 8 * hi));
        sc[nt] = WMMA_BF16(qfrag[kc], kfrag, sc[nt]);
      }
    }

    // ---- online softmax (rows live per lane-half; reduce over 16 lanes) ----
#pragma unroll
    for (int r = 0; r < 8; ++r) {
      float tmax = half_max16(fmaxf(sc[0][r], sc[1][r]));
      float mnew = fmaxf(mstate[r], tmax);
      float corr = __expf(mstate[r] - mnew);
      mstate[r] = mnew;
      float p0 = __expf(sc[0][r] - mnew);
      float p1 = __expf(sc[1][r] - mnew);
      lstate[r] = lstate[r] * corr + half_sum16(p0 + p1);
#pragma unroll
      for (int j = 0; j < 8; ++j) acc[j][r] *= corr;
      Ps[wv][r + 8 * hi][l16]      = (bf16)p0;
      Ps[wv][r + 8 * hi][l16 + 16] = (bf16)p1;
    }

    // ---- relayout P (C-layout -> A-layout) through per-wave LDS ----
    const bf16* prow = &Ps[wv][l16][0];
    v16bf pfrag = make_frag(*(const uint4*)(prow + 8 * hi),
                            *(const uint4*)(prow + 16 + 8 * hi));

    // ---- acc += P V ----
#pragma unroll
    for (int j = 0; j < 8; ++j) {
      const bf16* vp = &VsT[cur][j * 16 + l16][0];
      v16bf vfrag = make_frag(*(const uint4*)(vp + 8 * hi),
                              *(const uint4*)(vp + 16 + 8 * hi));
      acc[j] = WMMA_BF16(pfrag, vfrag, acc[j]);
    }

    async_wait0();
    __syncthreads();
  }

  // ---- finalize + write (b*S+s, h*DH+dh) bf16 for O-proj GEMM ----
#pragma unroll
  for (int j = 0; j < 8; ++j) {
    int col = h * 128 + j * 16 + l16;
#pragma unroll
    for (int r = 0; r < 8; ++r) {
      int srow = m0 + r + 8 * hi;
      attn[((size_t)b * 1024 + srow) * 2048 + col] =
          (bf16)(acc[j][r] / lstate[r]);
    }
  }
}

// ---------------------------------------------------------------------------
// Launch
// ---------------------------------------------------------------------------
extern "C" void kernel_launch(void* const* d_in, const int* in_sizes, int n_in,
                              void* d_out, int out_size, void* d_ws,
                              size_t ws_size, hipStream_t stream) {
  (void)in_sizes; (void)n_in; (void)out_size; (void)ws_size;

  const float* x       = (const float*)d_in[0];
  const float* pe      = (const float*)d_in[1];
  const float* w_qkv   = (const float*)d_in[2];
  const float* w_o     = (const float*)d_in[3];
  const float* q_scale = (const float*)d_in[4];
  const float* q_bias  = (const float*)d_in[5];
  const float* k_scale = (const float*)d_in[6];
  const float* k_bias  = (const float*)d_in[7];
  float* out = (float*)d_out;

  const size_t NX   = 4096ull * 2048;       // x / attn elements
  const size_t NWQ  = 2048ull * 6144;       // w_qkv elements
  const size_t NWO  = 2048ull * 2048;       // w_o elements
  const size_t NQKV = 4096ull * 6144;       // qkv elements
  const size_t NPK  = 64ull * 1024 * 128;   // packed per-tensor (B*H,S,DH)

  char* p = (char*)d_ws;
  bf16* x_bf    = (bf16*)p; p += NX * 2;
  bf16* wqkv_bf = (bf16*)p; p += NWQ * 2;
  bf16* wo_bf   = (bf16*)p; p += NWO * 2;
  bf16* qkv_bf  = (bf16*)p; p += NQKV * 2;
  bf16* Qp      = (bf16*)p; p += NPK * 2;
  bf16* Kp      = (bf16*)p; p += NPK * 2;
  bf16* Vp      = (bf16*)p; p += NPK * 2;
  bf16* attn_bf = (bf16*)p; p += NX * 2;    // ~168 MB total

  f32_to_bf16_kernel<<<2048, 256, 0, stream>>>(x, x_bf, NX);
  f32_to_bf16_kernel<<<2048, 256, 0, stream>>>(w_qkv, wqkv_bf, NWQ);
  f32_to_bf16_kernel<<<2048, 256, 0, stream>>>(w_o, wo_bf, NWO);

  // QKV projection: (4096x2048) @ (2048x6144) -> bf16
  gemm_wmma_bf16<false><<<dim3(6144 / 256, 4096 / 128), 256, 0, stream>>>(
      x_bf, wqkv_bf, (void*)qkv_bf, 4096, 6144, 2048);

  rmsnorm_rope_pack<<<4096, 256, 0, stream>>>(qkv_bf, pe, q_scale, q_bias,
                                              k_scale, k_bias, Qp, Kp, Vp);

  flash_attn_wmma<<<dim3(16, 64), 128, 0, stream>>>(Qp, Kp, Vp, attn_bf);

  // Output projection: (4096x2048) @ (2048x2048) -> fp32 out
  gemm_wmma_bf16<true><<<dim3(2048 / 256, 4096 / 128), 256, 0, stream>>>(
      attn_bf, wo_bf, (void*)out, 4096, 2048, 2048);
}
